// VisionMamba_54966991454799
// MI455X (gfx1250) — matrix operand order repaired and checked by reading, not compiled, not run on hardware
//
#include <hip/hip_runtime.h>
#include <hip/hip_bf16.h>
#include <math.h>

#define DEPTH   24
#define D_MODEL 192
#define D_INNER 384
#define D_STATE 16
#define DT_RANK 12
#define BATCH   64
#define SEQ     196
#define NTOK    (BATCH*SEQ)   /* 12544 tokens */
#define XDBL    44
#define XDBL_PAD 48
#define XZW     (2*D_INNER)   /* 768 */

typedef __attribute__((ext_vector_type(16))) __bf16 v16bf;
typedef __attribute__((ext_vector_type(8)))  __bf16 v8bf;
typedef __attribute__((ext_vector_type(8)))  float  v8f;

__device__ __forceinline__ float sigmoidf_(float x){ return 1.f/(1.f+__expf(-x)); }
__device__ __forceinline__ float softplusf_(float x){ return x>20.f ? x : log1pf(__expf(x)); }

// ---------------------------------------------------------------------------
// Weight conversion: fp32 -> bf16 (row-major N x K, contiguous)
// ---------------------------------------------------------------------------
__global__ void cvt_bf16(const float* __restrict__ s, __bf16* __restrict__ d, int n){
    int i = blockIdx.x*256 + threadIdx.x;
    if (i < n) d[i] = (__bf16)s[i];
}

// x_proj weights: (DEPTH,44,384) -> padded (DEPTH,48,384) bf16, zero rows 44..47
__global__ void xp_pad_bf16(const float* __restrict__ s, __bf16* __restrict__ d){
    int i = blockIdx.x*256 + threadIdx.x;
    if (i >= DEPTH*XDBL_PAD*D_INNER) return;
    int k = i % D_INNER;
    int n = (i / D_INNER) % XDBL_PAD;
    int layer = i / (D_INNER*XDBL_PAD);
    d[i] = (n < XDBL) ? (__bf16)s[((size_t)(layer*XDBL+n))*D_INNER + k] : (__bf16)0.f;
}

// ---------------------------------------------------------------------------
// Patch embed: 16x16 stride-16 conv -> hidden0[(b*196+l)*192 + d]
// ---------------------------------------------------------------------------
__global__ void patch_embed(const float* __restrict__ x, const float* __restrict__ pw,
                            const float* __restrict__ pb, float* __restrict__ h0){
    int idx = blockIdx.x*256 + threadIdx.x;
    if (idx >= NTOK*D_MODEL) return;
    int d = idx % D_MODEL;
    int l = (idx / D_MODEL) % SEQ;
    int b = idx / (D_MODEL*SEQ);
    int px = l % 14, py = l / 14;
    const float* xb = x + (size_t)b*3*224*224;
    const float* wd = pw + (size_t)d*3*256;
    float s = pb[d];
    for (int c = 0; c < 3; ++c) {
        const float* xp = xb + (size_t)c*224*224 + (py*16)*224 + px*16;
        const float* wc = wd + c*256;
        for (int ky = 0; ky < 16; ++ky) {
            #pragma unroll 4
            for (int kx = 0; kx < 16; ++kx)
                s += wc[ky*16+kx] * xp[ky*224+kx];
        }
    }
    h0[idx] = s;
}

// ---------------------------------------------------------------------------
// residual update + LayerNorm -> bf16 normed.  One wave32 per token row.
// ---------------------------------------------------------------------------
__global__ void resid_ln(float* __restrict__ residual, const float* __restrict__ hid,
                         const float* __restrict__ w, const float* __restrict__ bb,
                         __bf16* __restrict__ normed, int first){
    int row  = blockIdx.x*8 + (threadIdx.x >> 5);
    int lane = threadIdx.x & 31;
    size_t base = (size_t)row * D_MODEL;
    float v[6];
    float s = 0.f;
    #pragma unroll
    for (int i = 0; i < 6; ++i) {
        int e = lane + i*32;
        float xv = hid[base+e];
        if (!first) xv += residual[base+e];
        residual[base+e] = xv;
        v[i] = xv; s += xv;
    }
    #pragma unroll
    for (int m = 16; m >= 1; m >>= 1) s += __shfl_xor(s, m, 32);
    float mean = s * (1.f/D_MODEL);
    float q = 0.f;
    #pragma unroll
    for (int i = 0; i < 6; ++i) { float t = v[i]-mean; q += t*t; }
    #pragma unroll
    for (int m = 16; m >= 1; m >>= 1) q += __shfl_xor(q, m, 32);
    float rs = rsqrtf(q*(1.f/D_MODEL) + 1e-5f);
    #pragma unroll
    for (int i = 0; i < 6; ++i) {
        int e = lane + i*32;
        normed[base+e] = (__bf16)((v[i]-mean)*rs*w[e] + bb[e]);
    }
}

// ---------------------------------------------------------------------------
// WMMA GEMM, register-blocked in N:
//   C[M x Nstore] = A[M x K](bf16) * W[Npad x K](bf16)^T, f32 accumulate.
// grid = (Npad/(16*NT), M/64), block = 128 (4 waves).
// Each wave owns one 16-row M tile and NT 16-col N tiles: per 32-wide K step
// it loads ONE A fragment and NT B fragments -> NT back-to-back WMMAs on
// independent accumulators (no RAW hazard, feeds the XDL pipe).
// A fragment (16x32 bf16, ISA layout): lane m holds K 0-7|16-23 (lo half) or
// 8-15|24-31 (hi half).  B fragment: same K split as A, from row n.
// D layout: lane = n + 16*(m>=8), VGPR r = m&7.
// ---------------------------------------------------------------------------
template<int NT>
__global__ void gemm_bf16_wmma(const __bf16* __restrict__ A,
                               const __bf16* __restrict__ W,
                               float* __restrict__ C,
                               int M, int K, int Nstore){
    const int lane  = threadIdx.x & 31;
    const int wave  = threadIdx.x >> 5;
    const int mtile = blockIdx.y * 4 + wave;
    const int r     = lane & 15;
    const int half  = lane >> 4;

    const int m  = mtile*16 + r;
    const int n0 = (blockIdx.x*NT)*16 + r;

    const __bf16* ap = A + (size_t)m * K + half*8;
    const __bf16* wp = W + (size_t)n0 * K + half*8;
    const size_t  wstride = (size_t)16 * K;   // advance 16 weight rows per N tile

    v8f acc[NT];
    #pragma unroll
    for (int j = 0; j < NT; ++j) acc[j] = (v8f){};

    for (int k0 = 0; k0 < K; k0 += 32) {
        v8bf a_lo = *(const v8bf*)(ap + k0);
        v8bf a_hi = *(const v8bf*)(ap + k0 + 16);
        v16bf af = __builtin_shufflevector(a_lo, a_hi,
                     0,1,2,3,4,5,6,7,8,9,10,11,12,13,14,15);
        #pragma unroll
        for (int j = 0; j < NT; ++j) {
            const __bf16* wpj = wp + (size_t)j * wstride + k0;
            v8bf b_lo = *(const v8bf*)(wpj);
            v8bf b_hi = *(const v8bf*)(wpj + 16);
            v16bf bf = __builtin_shufflevector(b_lo, b_hi,
                         0,1,2,3,4,5,6,7,8,9,10,11,12,13,14,15);
            acc[j] = __builtin_amdgcn_wmma_f32_16x16x32_bf16(
                         false, af, false, bf, (short)0, acc[j], false, false);
        }
    }

    const int mbase = mtile*16 + half*8;
    #pragma unroll
    for (int j = 0; j < NT; ++j) {
        int n = n0 + j*16;
        if (n < Nstore) {
            float* cp = C + (size_t)mbase * Nstore + n;
            #pragma unroll
            for (int i = 0; i < 8; ++i) cp[(size_t)i * Nstore] = acc[j][i];
        }
    }
}

// ---------------------------------------------------------------------------
// Causal depthwise conv (kernel 4, left pad 3) + SiLU -> xc (f32 + bf16)
// ---------------------------------------------------------------------------
__global__ void conv_silu(const float* __restrict__ xz,
                          const float* __restrict__ cw, const float* __restrict__ cb,
                          float* __restrict__ xcf, __bf16* __restrict__ xcb){
    int idx = blockIdx.x*256 + threadIdx.x;
    if (idx >= NTOK*D_INNER) return;
    int d = idx % D_INNER;
    int l = (idx / D_INNER) % SEQ;
    int b = idx / (D_INNER*SEQ);
    float s = cb[d];
    #pragma unroll
    for (int j = 0; j < 4; ++j) {
        int ll = l - 3 + j;
        if (ll >= 0) s += cw[d*4+j] * xz[((size_t)(b*SEQ+ll))*XZW + d];
    }
    s = s * sigmoidf_(s);
    xcf[idx] = s;
    xcb[idx] = (__bf16)s;
}

// ---------------------------------------------------------------------------
// Fused selective scan: delta(softplus) / dA / dBx / recurrence / C-dot /
// + D skip / * silu(z).  One thread per (b,d) channel, L serial, x_dbl row
// staged in LDS and shared by the 128 channels of the block.
// grid = (D_INNER/128, BATCH), block = 128.
// ---------------------------------------------------------------------------
__global__ void scan_kernel(const float* __restrict__ xdbl,
                            const float* __restrict__ xc,
                            const float* __restrict__ xz,
                            const float* __restrict__ dtw,
                            const float* __restrict__ dtb,
                            const float* __restrict__ alog,
                            const float* __restrict__ Dp,
                            __bf16* __restrict__ ybf){
    const int d = blockIdx.x*128 + threadIdx.x;
    const int b = blockIdx.y;
    __shared__ float sd[XDBL];
    float A[D_STATE], h[D_STATE];
    #pragma unroll
    for (int n = 0; n < D_STATE; ++n) { A[n] = -__expf(alog[d*D_STATE+n]); h[n] = 0.f; }
    float wdt[DT_RANK];
    #pragma unroll
    for (int r = 0; r < DT_RANK; ++r) wdt[r] = dtw[d*DT_RANK+r];
    const float bias = dtb[d], dval = Dp[d];

    for (int l = 0; l < SEQ; ++l) {
        __syncthreads();
        if (threadIdx.x < XDBL)
            sd[threadIdx.x] = xdbl[((size_t)(b*SEQ+l))*XDBL + threadIdx.x];
        __syncthreads();
        float dt = bias;
        #pragma unroll
        for (int r = 0; r < DT_RANK; ++r) dt += sd[r]*wdt[r];
        float delta = softplusf_(dt);
        size_t tok = (size_t)(b*SEQ + l);
        float xcv = xc[tok*D_INNER + d];
        float dx = delta * xcv;
        float y = 0.f;
        #pragma unroll
        for (int n = 0; n < D_STATE; ++n) {
            h[n] = __expf(delta*A[n])*h[n] + dx*sd[DT_RANK+n];
            y += h[n]*sd[DT_RANK+D_STATE+n];
        }
        y += xcv * dval;
        float z = xz[tok*XZW + D_INNER + d];
        y *= z * sigmoidf_(z);
        ybf[tok*D_INNER + d] = (__bf16)y;
    }
}

// ---------------------------------------------------------------------------
// Final: residual += hidden, LN(normf), mean-pool over L.  One block per batch.
// ---------------------------------------------------------------------------
__global__ void final_ln_pool(const float* __restrict__ residual, const float* __restrict__ hid,
                              const float* __restrict__ w, const float* __restrict__ bb,
                              float* __restrict__ pooled){
    int b = blockIdx.x;
    int d = threadIdx.x;           // 0..191
    int lane = d & 31, wave = d >> 5;
    __shared__ float s1[6], s2[6];
    float acc = 0.f;
    const float wd = w[d], bd = bb[d];
    for (int l = 0; l < SEQ; ++l) {
        size_t base = ((size_t)(b*SEQ+l))*D_MODEL;
        float x = residual[base+d] + hid[base+d];
        float s = x, q = x*x;
        #pragma unroll
        for (int m = 16; m >= 1; m >>= 1) { s += __shfl_xor(s,m,32); q += __shfl_xor(q,m,32); }
        if (lane == 0) { s1[wave] = s; s2[wave] = q; }
        __syncthreads();
        float S = 0.f, Q = 0.f;
        #pragma unroll
        for (int i = 0; i < 6; ++i) { S += s1[i]; Q += s2[i]; }
        float mean = S*(1.f/D_MODEL);
        float var  = Q*(1.f/D_MODEL) - mean*mean;
        float rs   = rsqrtf(var + 1e-5f);
        acc += (x-mean)*rs*wd + bd;
        __syncthreads();
    }
    pooled[b*D_MODEL + d] = acc * (1.f/SEQ);
}

__global__ void head_kernel(const float* __restrict__ pooled, const float* __restrict__ hw,
                            const float* __restrict__ hb, float* __restrict__ out){
    int idx = blockIdx.x*256 + threadIdx.x;
    if (idx >= BATCH*1000) return;
    int o = idx % 1000, b = idx / 1000;
    float s = hb[o];
    const float* p  = pooled + b*D_MODEL;
    const float* wr = hw + (size_t)o*D_MODEL;
    #pragma unroll 4
    for (int dd = 0; dd < D_MODEL; ++dd) s += p[dd]*wr[dd];
    out[idx] = s;
}

// ---------------------------------------------------------------------------
extern "C" void kernel_launch(void* const* d_in, const int* in_sizes, int n_in,
                              void* d_out, int out_size, void* d_ws, size_t ws_size,
                              hipStream_t stream) {
    const float* x       = (const float*)d_in[0];
    const float* patch_w = (const float*)d_in[1];
    const float* patch_b = (const float*)d_in[2];
    const float* norm_w  = (const float*)d_in[3];
    const float* norm_b  = (const float*)d_in[4];
    const float* in_w    = (const float*)d_in[5];
    const float* conv_w  = (const float*)d_in[6];
    const float* conv_b  = (const float*)d_in[7];
    const float* xp_w    = (const float*)d_in[8];
    const float* dt_w    = (const float*)d_in[9];
    const float* dt_b    = (const float*)d_in[10];
    const float* a_log   = (const float*)d_in[11];
    const float* Dpv     = (const float*)d_in[12];
    const float* out_w   = (const float*)d_in[13];
    const float* normf_w = (const float*)d_in[14];
    const float* normf_b = (const float*)d_in[15];
    const float* head_w  = (const float*)d_in[16];
    const float* head_b  = (const float*)d_in[17];
    float* out = (float*)d_out;

    char* base = (char*)d_ws;
    size_t off = 0;
    auto alloc = [&](size_t bytes)->void* {
        void* p = base + off; off += (bytes + 255) & ~(size_t)255; return p;
    };

    float*  residual = (float*) alloc((size_t)NTOK*D_MODEL*4);
    float*  hidden   = (float*) alloc((size_t)NTOK*D_MODEL*4);
    float*  xz       = (float*) alloc((size_t)NTOK*XZW*4);
    float*  xcf      = (float*) alloc((size_t)NTOK*D_INNER*4);
    float*  xdbl     = (float*) alloc((size_t)NTOK*XDBL*4);
    float*  pooled   = (float*) alloc((size_t)BATCH*D_MODEL*4);
    __bf16* normedb  = (__bf16*)alloc((size_t)NTOK*D_MODEL*2);
    __bf16* xcb      = (__bf16*)alloc((size_t)NTOK*D_INNER*2);
    __bf16* yb       = (__bf16*)alloc((size_t)NTOK*D_INNER*2);
    __bf16* winb     = (__bf16*)alloc((size_t)DEPTH*XZW*D_MODEL*2);
    __bf16* woutb    = (__bf16*)alloc((size_t)DEPTH*D_MODEL*D_INNER*2);
    __bf16* wxpb     = (__bf16*)alloc((size_t)DEPTH*XDBL_PAD*D_INNER*2);

    int n1 = DEPTH*XZW*D_MODEL;
    cvt_bf16<<<(n1+255)/256,256,0,stream>>>(in_w, winb, n1);
    int n2 = DEPTH*D_MODEL*D_INNER;
    cvt_bf16<<<(n2+255)/256,256,0,stream>>>(out_w, woutb, n2);
    int n3 = DEPTH*XDBL_PAD*D_INNER;
    xp_pad_bf16<<<(n3+255)/256,256,0,stream>>>(xp_w, wxpb);

    patch_embed<<<(NTOK*D_MODEL+255)/256,256,0,stream>>>(x, patch_w, patch_b, hidden);

    for (int d = 0; d < DEPTH; ++d) {
        resid_ln<<<NTOK/8,256,0,stream>>>(residual, hidden,
            norm_w + d*D_MODEL, norm_b + d*D_MODEL, normedb, d == 0);

        // in_proj: N = 768 = 12 blocks * 4 tiles * 16
        gemm_bf16_wmma<4><<<dim3(XZW/64, NTOK/64),128,0,stream>>>(
            normedb, winb + (size_t)d*XZW*D_MODEL, xz, NTOK, D_MODEL, XZW);

        conv_silu<<<(NTOK*D_INNER+255)/256,256,0,stream>>>(
            xz, conv_w + d*D_INNER*4, conv_b + d*D_INNER, xcf, xcb);

        // x_proj: Npad = 48 = 1 block * 3 tiles * 16, store 44 cols
        gemm_bf16_wmma<3><<<dim3(1, NTOK/64),128,0,stream>>>(
            xcb, wxpb + (size_t)d*XDBL_PAD*D_INNER, xdbl, NTOK, D_INNER, XDBL);

        scan_kernel<<<dim3(D_INNER/128, BATCH),128,0,stream>>>(
            xdbl, xcf, xz,
            dt_w + d*D_INNER*DT_RANK, dt_b + d*D_INNER,
            a_log + d*D_INNER*D_STATE, Dpv + d*D_INNER, yb);

        // out_proj: N = 192 = 3 blocks * 4 tiles * 16
        gemm_bf16_wmma<4><<<dim3(D_MODEL/64, NTOK/64),128,0,stream>>>(
            yb, woutb + (size_t)d*D_MODEL*D_INNER, hidden, NTOK, D_INNER, D_MODEL);
    }

    final_ln_pool<<<BATCH, D_MODEL, 0, stream>>>(residual, hidden, normf_w, normf_b, pooled);
    head_kernel<<<(BATCH*1000+255)/256,256,0,stream>>>(pooled, head_w, head_b, out);
}
